// LabelCrossAttention_28896539968068
// MI455X (gfx1250) — matrix operand order repaired and checked by reading, not compiled, not run on hardware
//
#include <hip/hip_runtime.h>
#include <hip/hip_bf16.h>

#define B_ 2
#define S_ 4096
#define D_ 768
#define C_ 8192
#define DP_ (D_ / 2)   // 384 packed bf16 pairs along d
#define SP_ (S_ / 2)   // 2048 packed bf16 pairs along s
#define SBLK 128
#define NW 8           // waves per workgroup (wave32)
#define LN_EPS 1e-5f

typedef __attribute__((ext_vector_type(16))) __bf16 v16bf;
typedef __attribute__((ext_vector_type(8)))  float  v8f;

union Frag { unsigned int u[8]; v16bf v; };

__device__ __forceinline__ unsigned short f2bf(float f) {
  union { float f; unsigned int u; } a; a.f = f;
  unsigned int u = a.u;
  u += 0x7fffu + ((u >> 16) & 1u);   // round-to-nearest-even
  return (unsigned short)(u >> 16);
}
__device__ __forceinline__ unsigned int pack2(float lo, float hi) {
  return (unsigned int)f2bf(lo) | ((unsigned int)f2bf(hi) << 16);
}
__device__ __forceinline__ float rmax16(float x) {
  x = fmaxf(x, __shfl_xor(x, 1, 32));
  x = fmaxf(x, __shfl_xor(x, 2, 32));
  x = fmaxf(x, __shfl_xor(x, 4, 32));
  x = fmaxf(x, __shfl_xor(x, 8, 32));
  return x;
}
__device__ __forceinline__ float rsum16(float x) {
  x += __shfl_xor(x, 1, 32);
  x += __shfl_xor(x, 2, 32);
  x += __shfl_xor(x, 4, 32);
  x += __shfl_xor(x, 8, 32);
  return x;
}
__device__ __forceinline__ v8f vzero8() {
  v8f z;
#pragma unroll
  for (int i = 0; i < 8; ++i) z[i] = 0.f;
  return z;
}

// ---------------- pack kernels ----------------

// pack adjacent float pairs (inner dim) -> one dword of 2 bf16
__global__ void pack_pairs_kernel(const float* __restrict__ in,
                                  unsigned int* __restrict__ out, int n) {
  int i = blockIdx.x * blockDim.x + threadIdx.x;
  if (i < n) {
    const float2 f = ((const float2*)in)[i];
    out[i] = pack2(f.x, f.y);
  }
}

// W[e][d] (torch Linear weight) -> Wp[d/2][e] pair-packed over d (= B-matrix layout)
__global__ void pack_weightT_kernel(const float* __restrict__ W,
                                    unsigned int* __restrict__ out) {
  int i = blockIdx.x * blockDim.x + threadIdx.x;   // i = dp*D_ + e
  if (i < DP_ * D_) {
    int dp = i / D_, e = i - dp * D_;
    out[i] = pack2(W[e * D_ + 2 * dp], W[e * D_ + 2 * dp + 1]);
  }
}

// ---------------- K/V projection (WMMA bf16) ----------------
// block = 16 rows of x (one M-tile), 8 waves each produce a 16x96 slice of the
// 768-wide output.  mode 0: K -> Kp[b][dp][s] (transposed pair-packed)
//                   mode 1: V=xW^T+bv -> Vp[b][sp][d] (s-pair-packed)
__global__ __launch_bounds__(256) void proj_kernel(
    const float* __restrict__ x, const unsigned int* __restrict__ Wp,
    const float* __restrict__ bias, unsigned int* __restrict__ Kp,
    unsigned int* __restrict__ Vp, int mode) {
  __shared__ unsigned int Xs[16][DP_ + 1];   // +1 dword pad: bank-conflict-free A reads
  const int tid = threadIdx.x;
  const int w = tid >> 5, lane = tid & 31, ln = lane & 15, g = lane >> 4;
  const int r0 = blockIdx.x * 16;            // flat row in [0, B*S)

  for (int i = tid; i < 16 * DP_; i += 256) {
    int rr = i / DP_, cc = i - rr * DP_;
    const float2 f = *(const float2*)(x + (size_t)(r0 + rr) * D_ + 2 * cc);
    Xs[rr][cc] = pack2(f.x, f.y);
  }
  __syncthreads();

  const int e0 = w * 96;
  v8f acc[6];
#pragma unroll
  for (int t = 0; t < 6; ++t) acc[t] = vzero8();

  for (int kc = 0; kc < 24; ++kc) {          // K-dim 768 = 24 x 32
    Frag a;
#pragma unroll
    for (int v = 0; v < 8; ++v)
      a.u[v] = Xs[ln][kc * 16 + g * 8 + v];
    const unsigned int* Wb = Wp + (size_t)(kc * 16 + g * 8) * D_ + e0 + ln;
#pragma unroll
    for (int t = 0; t < 6; ++t) {
      Frag bf;
#pragma unroll
      for (int v = 0; v < 8; ++v) bf.u[v] = Wb[(size_t)v * D_ + t * 16];
      acc[t] = __builtin_amdgcn_wmma_f32_16x16x32_bf16(
          false, a.v, false, bf.v, (short)0, acc[t], false, false);
    }
  }

  const int b   = r0 >> 12;        // r0 / S_
  const int s0l = r0 & (S_ - 1);
  if (mode == 0) {
#pragma unroll
    for (int t = 0; t < 6; ++t) {
      const int e = e0 + t * 16 + ln;
#pragma unroll
      for (int v = 0; v < 8; ++v) {
        float val = acc[t][v];
        float nb  = __shfl_xor(val, 1, 32);  // neighbor column e^1
        if ((ln & 1) == 0) {
          const int s = s0l + v + 8 * g;
          Kp[(size_t)b * DP_ * S_ + (size_t)(e >> 1) * S_ + s] = pack2(val, nb);
        }
      }
    }
  } else {
#pragma unroll
    for (int t = 0; t < 6; ++t) {
      const int e  = e0 + t * 16 + ln;
      const float bb = bias[e];
#pragma unroll
      for (int v = 0; v < 8; v += 2) {       // row pairs (v, v+1) live in-lane
        const float v0 = acc[t][v] + bb;
        const float v1 = acc[t][v + 1] + bb;
        const int sp = (s0l + v + 8 * g) >> 1;
        Vp[(size_t)b * SP_ * D_ + (size_t)sp * D_ + e] = pack2(v0, v1);
      }
    }
  }
}

// ---------------- fused flash label-attention + LayerNorm + head ----------------
__global__ __launch_bounds__(256) void flash_kernel(
    const unsigned int* __restrict__ Qp, const unsigned int* __restrict__ Kp,
    const unsigned int* __restrict__ Vp, const int* __restrict__ amask,
    const float* __restrict__ gamma, const float* __restrict__ beta,
    const float* __restrict__ wout, const float* __restrict__ bout,
    float* __restrict__ out) {
  __shared__ unsigned int Qs[16][DP_ + 1];                 // 16 labels x 384 pairs
  __shared__ __align__(16) unsigned short Ps[16][SBLK + 4]; // probs, pad vs bank hits
  __shared__ float redA[16][NW];
  __shared__ float redB[16][NW];
  __shared__ float mrow[16], lrow[16], arow[16];

  const int tid = threadIdx.x;
  const int w = tid >> 5, lane = tid & 31, ln = lane & 15, g = lane >> 4;
  const int bc = blockIdx.x;
  const int b  = bc >> 9;                    // 512 c-tiles per batch
  const int c0 = (bc & 511) * 16;

  for (int i = tid; i < 16 * DP_; i += 256) {
    int rr = i / DP_, cc = i - rr * DP_;
    Qs[rr][cc] = Qp[(size_t)(c0 + rr) * DP_ + cc];
  }
  if (tid < 16) { mrow[tid] = -__builtin_inff(); lrow[tid] = 0.f; }

  v8f O[6];
#pragma unroll
  for (int t = 0; t < 6; ++t) O[t] = vzero8();
  __syncthreads();

  const unsigned int* Kb = Kp + (size_t)b * DP_ * S_;
  const unsigned int* Vb = Vp + (size_t)b * SP_ * D_;
  const int* mb = amask + b * S_;

  for (int sb = 0; sb < S_ / SBLK; ++sb) {
    const int s0   = sb * SBLK;
    const int scol = s0 + w * 16 + ln;       // this lane's score column

    // (1) scores: 16 labels x 16 keys per wave, 24 WMMAs over D=768
    v8f Sc = vzero8();
    for (int kc = 0; kc < 24; ++kc) {
      Frag a, bf;
#pragma unroll
      for (int v = 0; v < 8; ++v)
        a.u[v] = Qs[ln][kc * 16 + g * 8 + v];
      const unsigned int* kp = Kb + (size_t)(kc * 16 + g * 8) * S_ + scol;
#pragma unroll
      for (int v = 0; v < 8; ++v) bf.u[v] = kp[(size_t)v * S_];
      Sc = __builtin_amdgcn_wmma_f32_16x16x32_bf16(
          false, a.v, false, bf.v, (short)0, Sc, false, false);
    }
    // (2) additive mask: lane owns a single s column
    if (mb[scol] == 0) {
#pragma unroll
      for (int v = 0; v < 8; ++v) Sc[v] = -__builtin_inff();
    }
    // (3) per-row running max (cross-lane then cross-wave)
#pragma unroll
    for (int v = 0; v < 8; ++v) {
      float mx = rmax16(Sc[v]);
      if (ln == 0) redA[v + 8 * g][w] = mx;
    }
    __syncthreads();
    if (tid < 16) {
      float bm = redA[tid][0];
#pragma unroll
      for (int j = 1; j < NW; ++j) bm = fmaxf(bm, redA[tid][j]);
      const float mo = mrow[tid];
      const float mn = fmaxf(mo, bm);
      arow[tid] = (mo > -__builtin_inff()) ? __expf(mo - mn) : 0.f;
      mrow[tid] = mn;
    }
    __syncthreads();
    // (4) rescale O, P = exp(Sc - m_new), stash P (bf16) + row sums
    float al8[8];
#pragma unroll
    for (int v = 0; v < 8; ++v) al8[v] = arow[v + 8 * g];
#pragma unroll
    for (int t = 0; t < 6; ++t)
#pragma unroll
      for (int v = 0; v < 8; ++v) O[t][v] *= al8[v];
#pragma unroll
    for (int v = 0; v < 8; ++v) {
      const int row = v + 8 * g;
      const float mn = mrow[row];
      const float p = (Sc[v] > -__builtin_inff()) ? __expf(Sc[v] - mn) : 0.f;
      Ps[row][w * 16 + ln] = f2bf(p);
      const float ps = rsum16(p);
      if (ln == 0) redB[row][w] = ps;
    }
    __syncthreads();
    if (tid < 16) {
      float sum = 0.f;
#pragma unroll
      for (int j = 0; j < NW; ++j) sum += redB[tid][j];
      lrow[tid] = lrow[tid] * arow[tid] + sum;
    }
    // (5) O += P x V  (wave owns d-slice of 96: 6 N-tiles x 4 K-chunks)
    const int d0 = w * 96;
#pragma unroll
    for (int kc2 = 0; kc2 < 4; ++kc2) {
      Frag a;
#pragma unroll
      for (int v = 0; v < 8; ++v) {
        const int k0 = kc2 * 32 + g * 16 + 2 * v;
        a.u[v] = *(const unsigned int*)&Ps[ln][k0];
      }
      const unsigned int* vp =
          Vb + (size_t)((s0 >> 1) + kc2 * 16 + g * 8) * D_ + d0 + ln;
#pragma unroll
      for (int t = 0; t < 6; ++t) {
        Frag bf;
#pragma unroll
        for (int v = 0; v < 8; ++v) bf.u[v] = vp[(size_t)v * D_ + t * 16];
        O[t] = __builtin_amdgcn_wmma_f32_16x16x32_bf16(
            false, a.v, false, bf.v, (short)0, O[t], false, false);
      }
    }
  }

  // ---- epilogue: y = O/l, LayerNorm over D, dot with w_out ----
  __syncthreads();
  float rl8[8];
#pragma unroll
  for (int v = 0; v < 8; ++v) rl8[v] = 1.f / lrow[v + 8 * g];
#pragma unroll
  for (int v = 0; v < 8; ++v) {
    float s1 = 0.f, s2 = 0.f;
#pragma unroll
    for (int t = 0; t < 6; ++t) {
      const float y = O[t][v] * rl8[v];
      s1 += y; s2 += y * y;
    }
    s1 = rsum16(s1); s2 = rsum16(s2);
    if (ln == 0) { redA[v + 8 * g][w] = s1; redB[v + 8 * g][w] = s2; }
  }
  __syncthreads();
  if (tid < 16) {
    float sa = 0.f, sq = 0.f;
#pragma unroll
    for (int j = 0; j < NW; ++j) { sa += redA[tid][j]; sq += redB[tid][j]; }
    const float mu  = sa * (1.f / D_);
    const float var = sq * (1.f / D_) - mu * mu;
    mrow[tid] = mu;
    arow[tid] = rsqrtf(var + LN_EPS);
  }
  __syncthreads();
  float g6[6], be6[6], wo6[6];
#pragma unroll
  for (int t = 0; t < 6; ++t) {
    const int d = w * 96 + t * 16 + ln;
    g6[t] = gamma[d]; be6[t] = beta[d]; wo6[t] = wout[d];
  }
#pragma unroll
  for (int v = 0; v < 8; ++v) {
    const int row = v + 8 * g;
    const float mu = mrow[row], inv = arow[row];
    float accz = 0.f;
#pragma unroll
    for (int t = 0; t < 6; ++t) {
      const float y = O[t][v] * rl8[v];
      accz += ((y - mu) * inv * g6[t] + be6[t]) * wo6[t];
    }
    accz = rsum16(accz);
    if (ln == 0) redA[row][w] = accz;
  }
  __syncthreads();
  if (tid < 16) {
    float s = 0.f;
#pragma unroll
    for (int j = 0; j < NW; ++j) s += redA[tid][j];
    out[b * C_ + c0 + tid] = s + bout[0];
  }
}

// ---------------- host launch ----------------
extern "C" void kernel_launch(void* const* d_in, const int* in_sizes, int n_in,
                              void* d_out, int out_size, void* d_ws, size_t ws_size,
                              hipStream_t stream) {
  const float* x  = (const float*)d_in[0];
  const int*   am = (const int*)d_in[1];
  const float* Wk = (const float*)d_in[2];
  const float* Wv = (const float*)d_in[3];
  const float* bv = (const float*)d_in[4];
  const float* Q  = (const float*)d_in[5];
  const float* ga = (const float*)d_in[6];
  const float* be = (const float*)d_in[7];
  const float* wo = (const float*)d_in[8];
  const float* bo = (const float*)d_in[9];
  float* out = (float*)d_out;

  // workspace carve-up (~40 MB): bf16 pair-packed operands
  unsigned int* p   = (unsigned int*)d_ws;
  unsigned int* Qp  = p; p += (size_t)C_ * DP_;        // [C][D/2]
  unsigned int* Wkp = p; p += (size_t)DP_ * D_;        // [D/2][D]
  unsigned int* Wvp = p; p += (size_t)DP_ * D_;
  unsigned int* Kp  = p; p += (size_t)B_ * DP_ * S_;   // [B][D/2][S]
  unsigned int* Vp  = p; p += (size_t)B_ * SP_ * D_;   // [B][S/2][D]

  pack_pairs_kernel<<<(C_ * DP_) / 256, 256, 0, stream>>>(Q, Qp, C_ * DP_);
  pack_weightT_kernel<<<(DP_ * D_) / 256, 256, 0, stream>>>(Wk, Wkp);
  pack_weightT_kernel<<<(DP_ * D_) / 256, 256, 0, stream>>>(Wv, Wvp);
  proj_kernel<<<(B_ * S_) / 16, 256, 0, stream>>>(x, Wkp, bv, Kp, Vp, 0);
  proj_kernel<<<(B_ * S_) / 16, 256, 0, stream>>>(x, Wvp, bv, Kp, Vp, 1);
  flash_kernel<<<B_ * (C_ / 16), 256, 0, stream>>>(Qp, Kp, Vp, am, ga, be, wo, bo, out);
}
